// SelfAttention_25065429139771
// MI455X (gfx1250) — compile-verified, matrix operand order checked by reading, not run on hardware
//
#include <hip/hip_runtime.h>
#include <hip/hip_bf16.h>

// Problem constants (fixed by the reference instance)
#define BATCH 2
#define SEQ   2048
#define EMB   2048
#define HEADS 16
#define HDIM  128            // EMB / HEADS
#define MROWS (BATCH * SEQ)  // 4096
#define N_QKV (3 * EMB)      // 6144

typedef __attribute__((ext_vector_type(16))) __bf16 v16bf;
typedef __attribute__((ext_vector_type(8)))  float  v8f;
typedef int v4i __attribute__((vector_size(16)));

typedef unsigned short u16;

__device__ inline u16 f2bf(float f) {
    union { float f; unsigned u; } v; v.f = f;
    unsigned r = v.u + 0x7FFFu + ((v.u >> 16) & 1u);   // round-to-nearest-even
    return (u16)(r >> 16);
}
__device__ inline float bf2f(u16 b) {
    return __uint_as_float(((unsigned)b) << 16);
}

__device__ inline v8f wmma_bf16(v16bf a, v16bf b, v8f c) {
    // (neg_a, A, neg_b, B, c_mod, C, reuse_a, reuse_b)
    return __builtin_amdgcn_wmma_f32_16x16x32_bf16(false, a, false, b, (short)0, c, false, false);
}

__device__ inline float rmax16(float v) {
#pragma unroll
    for (int m = 1; m < 16; m <<= 1) v = fmaxf(v, __shfl_xor(v, m, 32));
    return v;
}
__device__ inline float rsum16(float v) {
#pragma unroll
    for (int m = 1; m < 16; m <<= 1) v += __shfl_xor(v, m, 32);
    return v;
}

// ---- CDNA5 async global->LDS helpers (ASYNCcnt path), guarded ------------
__device__ inline void async_copy16(u16* lds_dst, const u16* gsrc) {
#if defined(__AMDGCN__) && __has_builtin(__builtin_amdgcn_global_load_async_to_lds_b128)
    typedef __attribute__((address_space(1))) v4i* gp_t;   // global (AS1)
    typedef __attribute__((address_space(3))) v4i* lp_t;   // LDS (AS3)
    // generic LDS pointer: low 32 bits are the LDS byte address (aperture rules)
    __builtin_amdgcn_global_load_async_to_lds_b128(
        (gp_t)(unsigned long long)gsrc,
        (lp_t)(unsigned)(unsigned long long)lds_dst, 0, 0);
#else
    *reinterpret_cast<uint4*>(lds_dst) = *reinterpret_cast<const uint4*>(gsrc);
#endif
}

template <int N>
__device__ inline void wait_async() {
#if defined(__AMDGCN__) && __has_builtin(__builtin_amdgcn_s_wait_asynccnt)
    __builtin_amdgcn_s_wait_asynccnt(N);
#elif defined(__AMDGCN__)
    if (N == 8)      asm volatile("s_wait_asynccnt 8" ::: "memory");
    else             asm volatile("s_wait_asynccnt 0" ::: "memory");
#endif
}

// ---------------------------------------------------------------------------
// Kernel 1: fp32 -> bf16 conversion (vectorized)
// ---------------------------------------------------------------------------
__global__ __launch_bounds__(256) void cvt_f32_bf16(const float* __restrict__ in,
                                                    u16* __restrict__ out, int n) {
    int i = (blockIdx.x * 256 + threadIdx.x) * 4;
    if (i + 3 < n) {
        float4 v = *reinterpret_cast<const float4*>(in + i);
        ushort4 o;
        o.x = f2bf(v.x); o.y = f2bf(v.y); o.z = f2bf(v.z); o.w = f2bf(v.w);
        *reinterpret_cast<ushort4*>(out + i) = o;
    } else {
        for (int j = i; j < n; ++j) out[j] = f2bf(in[j]);
    }
}

// ---------------------------------------------------------------------------
// Kernel 2/5: tiled WMMA GEMM  C[M,N] = A[M,K] * B[N,K]^T  (bf16 in, bf16/f32 out)
// 256 threads = 8 waves; block tile 128x128; K-step 64; double-buffered LDS
// filled with GLOBAL_LOAD_ASYNC_TO_LDS_B128 (ASYNCcnt-pipelined).
// Wave w: rows (w&3)*32..+32, cols (w>>2)*64..+64  -> 2x4 accumulators,
// 16 WMMAs per stage per wave.
// ---------------------------------------------------------------------------
template <int OUT_F32>
__global__ __launch_bounds__(256) void gemm_wmma(const u16* __restrict__ A,
                                                 const u16* __restrict__ Bm,
                                                 void* __restrict__ Cv,
                                                 int M, int N, int K) {
    __shared__ u16 As[2][128 * 64];   // 16 KB per buffer
    __shared__ u16 Bs[2][128 * 64];

    const int tid   = threadIdx.x;
    const int w     = tid >> 5;
    const int lane  = tid & 31;
    const int l15   = lane & 15;
    const int lhalf = lane >> 4;
    const int m0    = blockIdx.y * 128;
    const int n0    = blockIdx.x * 128;
    const int wm    = (w & 3) * 32;
    const int wn    = (w >> 2) * 64;

    v8f acc[2][4];
#pragma unroll
    for (int mi = 0; mi < 2; ++mi)
#pragma unroll
        for (int ni = 0; ni < 4; ++ni)
            acc[mi][ni] = v8f{0.f, 0.f, 0.f, 0.f, 0.f, 0.f, 0.f, 0.f};

    // stage issue: 128 rows x 64 halves = 1024 16B-chunks per tile; 4 per thread
    auto issue_stage = [&](int stg) {
        u16* abuf = &As[stg & 1][0];
        u16* bbuf = &Bs[stg & 1][0];
        const int kt = stg * 64;
#pragma unroll
        for (int i = 0; i < 4; ++i) {
            int c = tid + i * 256;
            int row = c >> 3, cpos = (c & 7) * 8;
            async_copy16(&abuf[row * 64 + cpos], &A[(size_t)(m0 + row) * K + kt + cpos]);
            async_copy16(&bbuf[row * 64 + cpos], &Bm[(size_t)(n0 + row) * K + kt + cpos]);
        }
    };

    const int nst = K / 64;
    issue_stage(0);
    for (int s = 0; s < nst; ++s) {
        if (s + 1 < nst) {
            issue_stage(s + 1);   // 8 newer async fills in flight
            wait_async<8>();      // wait only for stage-s fills (older 8)
        } else {
            wait_async<0>();
        }
        __syncthreads();          // all waves' stage-s data visible

        const u16* abuf = &As[s & 1][0];
        const u16* bbuf = &Bs[s & 1][0];
#pragma unroll
        for (int kc = 0; kc < 64; kc += 32) {
            v16bf af[2], bfv[4];
#pragma unroll
            for (int mi = 0; mi < 2; ++mi)
                af[mi] = *reinterpret_cast<const v16bf*>(
                    &abuf[(wm + mi * 16 + l15) * 64 + kc + lhalf * 16]);
#pragma unroll
            for (int ni = 0; ni < 4; ++ni)
                bfv[ni] = *reinterpret_cast<const v16bf*>(
                    &bbuf[(wn + ni * 16 + l15) * 64 + kc + lhalf * 16]);
#pragma unroll
            for (int mi = 0; mi < 2; ++mi)
#pragma unroll
                for (int ni = 0; ni < 4; ++ni)
                    acc[mi][ni] = wmma_bf16(af[mi], bfv[ni], acc[mi][ni]);
        }
        __syncthreads();          // buffer may be overwritten by stage s+2
    }

    // Epilogue: C layout row = base + r + 8*lhalf, col = base + l15
#pragma unroll
    for (int mi = 0; mi < 2; ++mi) {
#pragma unroll
        for (int ni = 0; ni < 4; ++ni) {
#pragma unroll
            for (int r = 0; r < 8; ++r) {
                int row = m0 + wm + mi * 16 + r + 8 * lhalf;
                int col = n0 + wn + ni * 16 + l15;
                float v = acc[mi][ni][r];
                if (OUT_F32)
                    reinterpret_cast<float*>(Cv)[(size_t)row * N + col] = v;
                else
                    reinterpret_cast<u16*>(Cv)[(size_t)row * N + col] = f2bf(v);
            }
        }
    }
}

// ---------------------------------------------------------------------------
// Kernel 3: RoPE + split.  qkv[(b,s), 3E] (bf16) ->
//   q,k: (B,H,S,D) bf16 with RoPE;  v: transposed (B,H,D,S) bf16.
// One thread per (b,s,h,pair i), i in [0,64).
// ---------------------------------------------------------------------------
__global__ __launch_bounds__(256) void rope_split(const u16* __restrict__ qkv,
                                                  u16* __restrict__ q,
                                                  u16* __restrict__ k,
                                                  u16* __restrict__ vT) {
    int idx = blockIdx.x * 256 + threadIdx.x;   // 2^22 total
    int i = idx & 63;
    int h = (idx >> 6) & 15;
    int s = (idx >> 10) & 2047;
    int b = idx >> 21;

    size_t mrow = ((size_t)b * SEQ + s) * (size_t)N_QKV;
    int col = h * HDIM + 2 * i;

    float q1 = bf2f(qkv[mrow + col]),             q2 = bf2f(qkv[mrow + col + 1]);
    float k1 = bf2f(qkv[mrow + EMB + col]),       k2 = bf2f(qkv[mrow + EMB + col + 1]);
    float v1 = bf2f(qkv[mrow + 2 * EMB + col]),   v2 = bf2f(qkv[mrow + 2 * EMB + col + 1]);

    // inv_freq = 10000^(-i/64) = exp(-i * ln(10000)/64)
    float theta = (float)s * __expf(-(float)i * 0.14391156f);
    float cs = __cosf(theta), sn = __sinf(theta);

    int bh = b * HEADS + h;
    size_t qo = ((size_t)bh * SEQ + s) * HDIM + 2 * i;
    q[qo]     = f2bf(q1 * cs - q2 * sn);
    q[qo + 1] = f2bf(q1 * sn + q2 * cs);
    k[qo]     = f2bf(k1 * cs - k2 * sn);
    k[qo + 1] = f2bf(k1 * sn + k2 * cs);

    size_t vo = ((size_t)bh * HDIM + 2 * i) * SEQ + s;
    vT[vo]       = f2bf(v1);
    vT[vo + SEQ] = f2bf(v2);
}

// ---------------------------------------------------------------------------
// Kernel 4: flash attention (causal).  grid = (S/128, B*H), 256 thr = 8 waves.
// Each wave owns 16 q rows; iterates 32-wide k tiles with online softmax.
// Output attn in (B,S,E) bf16 for the final projection GEMM.
// ---------------------------------------------------------------------------
__global__ __launch_bounds__(256) void flash_attn(const u16* __restrict__ q,
                                                  const u16* __restrict__ k,
                                                  const u16* __restrict__ vT,
                                                  u16* __restrict__ attn) {
    __shared__ u16 Plds[8][16][32];  // per-wave P staging (C-layout -> A-layout)

    const int tid   = threadIdx.x;
    const int w     = tid >> 5;
    const int lane  = tid & 31;
    const int l15   = lane & 15;
    const int lhalf = lane >> 4;
    const int bh    = blockIdx.y;          // 0..31
    const int b     = bh >> 4;
    const int h     = bh & 15;
    const int qr0   = blockIdx.x * 128 + w * 16;
    const float scale = 0.08838834764831845f;  // 1/sqrt(128)

    const u16* qb = q  + (size_t)bh * SEQ * HDIM;
    const u16* kb = k  + (size_t)bh * SEQ * HDIM;
    const u16* vb = vT + (size_t)bh * HDIM * SEQ;

    // Q fragments (16 rows x 128 d), 4 K-chunks of 32
    v16bf qf[4];
#pragma unroll
    for (int c = 0; c < 4; ++c)
        qf[c] = *reinterpret_cast<const v16bf*>(&qb[(size_t)(qr0 + l15) * HDIM + c * 32 + lhalf * 16]);

    v8f o[8];
#pragma unroll
    for (int c = 0; c < 8; ++c) o[c] = v8f{0.f, 0.f, 0.f, 0.f, 0.f, 0.f, 0.f, 0.f};
    float mrow[8], lrow[8];
#pragma unroll
    for (int r = 0; r < 8; ++r) { mrow[r] = -1e30f; lrow[r] = 0.f; }

    const int jend = qr0 + 16;                 // rows qr0..qr0+15 need cols <= qr0+15
    for (int j = 0; j < jend; j += 32) {
        // prefetch next k/v tile (lowers to global_prefetch_b8)
        if (j + 32 < jend) {
            __builtin_prefetch(&kb[(size_t)(j + 32 + l15) * HDIM], 0, 1);
            __builtin_prefetch(&vb[(size_t)(lane * 4) * SEQ + j + 32], 0, 1);
        }

        // ---- scores: 16 x 32, two 16-col WMMA chains over D ----
        v8f sc0 = v8f{0.f, 0.f, 0.f, 0.f, 0.f, 0.f, 0.f, 0.f};
        v8f sc1 = sc0;
        int kp0 = j + l15;
        int kp1 = j + 16 + l15; if (kp1 >= SEQ) kp1 = SEQ - 1;   // masked anyway
#pragma unroll
        for (int c = 0; c < 4; ++c) {
            v16bf kf0 = *reinterpret_cast<const v16bf*>(&kb[(size_t)kp0 * HDIM + c * 32 + lhalf * 16]);
            v16bf kf1 = *reinterpret_cast<const v16bf*>(&kb[(size_t)kp1 * HDIM + c * 32 + lhalf * 16]);
            sc0 = wmma_bf16(qf[c], kf0, sc0);
            sc1 = wmma_bf16(qf[c], kf1, sc1);
        }

        // ---- mask + online softmax (rows r+8*lhalf, cols l15 / 16+l15) ----
        float p0[8], p1[8], fac[8];
#pragma unroll
        for (int r = 0; r < 8; ++r) {
            int qrow = qr0 + r + 8 * lhalf;
            float s0 = sc0[r] * scale;
            float s1 = sc1[r] * scale;
            if (j + l15 > qrow)      s0 = -1e30f;
            if (j + 16 + l15 > qrow) s1 = -1e30f;
            float rm = rmax16(fmaxf(s0, s1));
            float mn = fmaxf(mrow[r], rm);
            float e0 = __expf(s0 - mn);
            float e1 = __expf(s1 - mn);
            float f  = __expf(mrow[r] - mn);
            lrow[r] = lrow[r] * f + rsum16(e0 + e1);
            mrow[r] = mn;
            fac[r] = f; p0[r] = e0; p1[r] = e1;
        }
#pragma unroll
        for (int c = 0; c < 8; ++c)
#pragma unroll
            for (int r = 0; r < 8; ++r) o[c][r] *= fac[r];

        // ---- stage P (C-layout) into LDS, reread as A fragment ----
#pragma unroll
        for (int r = 0; r < 8; ++r) {
            int row = r + 8 * lhalf;
            Plds[w][row][l15]      = f2bf(p0[r]);
            Plds[w][row][16 + l15] = f2bf(p1[r]);
        }
        asm volatile("s_wait_dscnt 0" ::: "memory");
        v16bf pa = *reinterpret_cast<const v16bf*>(&Plds[w][l15][lhalf * 16]);

        // ---- O += P (16x32) * V (32 x 128) ----
        int vs = j + lhalf * 16; if (vs + 16 > SEQ) vs = SEQ - 16;  // masked anyway
#pragma unroll
        for (int c = 0; c < 8; ++c) {
            v16bf vf = *reinterpret_cast<const v16bf*>(&vb[(size_t)(c * 16 + l15) * SEQ + vs]);
            o[c] = wmma_bf16(pa, vf, o[c]);
        }
    }

    // ---- normalize + write (B,S,E) bf16 ----
#pragma unroll
    for (int r = 0; r < 8; ++r) lrow[r] = 1.f / lrow[r];
#pragma unroll
    for (int c = 0; c < 8; ++c) {
#pragma unroll
        for (int r = 0; r < 8; ++r) {
            int s = qr0 + r + 8 * lhalf;
            int e = h * HDIM + c * 16 + l15;
            attn[((size_t)b * SEQ + s) * EMB + e] = f2bf(o[c][r] * lrow[r]);
        }
    }
}

// ---------------------------------------------------------------------------
extern "C" void kernel_launch(void* const* d_in, const int* in_sizes, int n_in,
                              void* d_out, int out_size, void* d_ws, size_t ws_size,
                              hipStream_t stream) {
    const float* x     = (const float*)d_in[0];   // (2,2048,2048)
    const float* w_qkv = (const float*)d_in[1];   // (6144,2048)
    const float* w_o   = (const float*)d_in[2];   // (2048,2048)
    float* out = (float*)d_out;

    // workspace layout (bf16 elements)
    u16* x_bf    = (u16*)d_ws;
    u16* wqkv_bf = x_bf    + (size_t)MROWS * EMB;       // 8 M elems
    u16* wo_bf   = wqkv_bf + (size_t)N_QKV * EMB;       // 12 M
    u16* qkv_bf  = wo_bf   + (size_t)EMB * EMB;         // 4 M
    u16* q_bf    = qkv_bf  + (size_t)MROWS * N_QKV;     // 24 M
    u16* k_bf    = q_bf    + (size_t)MROWS * EMB;
    u16* vT_bf   = k_bf    + (size_t)MROWS * EMB;
    u16* attn_bf = vT_bf   + (size_t)MROWS * EMB;

    const int nx = MROWS * EMB, nq = N_QKV * EMB, no = EMB * EMB;
    cvt_f32_bf16<<<(nx / 4 + 255) / 256, 256, 0, stream>>>(x, x_bf, nx);
    cvt_f32_bf16<<<(nq / 4 + 255) / 256, 256, 0, stream>>>(w_qkv, wqkv_bf, nq);
    cvt_f32_bf16<<<(no / 4 + 255) / 256, 256, 0, stream>>>(w_o, wo_bf, no);

    // qkv = x * w_qkv^T : M=4096, N=6144, K=2048
    gemm_wmma<0><<<dim3(N_QKV / 128, MROWS / 128), 256, 0, stream>>>(
        x_bf, wqkv_bf, qkv_bf, MROWS, N_QKV, EMB);

    // RoPE + split: 2^22 threads
    rope_split<<<(1 << 22) / 256, 256, 0, stream>>>(qkv_bf, q_bf, k_bf, vT_bf);

    // flash attention
    flash_attn<<<dim3(SEQ / 128, BATCH * HEADS), 256, 0, stream>>>(q_bf, k_bf, vT_bf, attn_bf);

    // out = attn * w_o^T : M=4096, N=2048, K=2048, fp32 out
    gemm_wmma<1><<<dim3(EMB / 128, MROWS / 128), 256, 0, stream>>>(
        attn_bf, wo_bf, out, MROWS, EMB, EMB);
}
